// GPT_61512521613735
// MI455X (gfx1250) — matrix operand order repaired and checked by reading, not compiled, not run on hardware
//
#include <hip/hip_runtime.h>
#include <hip/hip_bf16.h>

#define D    768
#define HNUM 12
#define NBLK 12
#define LSEQ 1024
#define VOCAB 50257
#define VPAD  50304          // 786 * 64
#define LN_EPS 1e-5f

typedef __attribute__((ext_vector_type(16))) __bf16 v16bf;
typedef __attribute__((ext_vector_type(8)))  __bf16 v8bf;
typedef __attribute__((ext_vector_type(8)))  float  v8f;
typedef __attribute__((ext_vector_type(8)))  float  f32x8;

__device__ inline v8f vzero8() {
  v8f v;
#pragma unroll
  for (int i = 0; i < 8; ++i) v[i] = 0.0f;
  return v;
}

__device__ inline v16bf cat8(v8bf lo, v8bf hi) {
  return __builtin_shufflevector(lo, hi, 0,1,2,3,4,5,6,7,8,9,10,11,12,13,14,15);
}

// A-fragment from bf16 row-major source: ptr already points at row*K + k0 + hi*16
__device__ inline v16bf load_a_bf16(const __bf16* p) {
  v8bf lo = *(const v8bf*)(p);
  v8bf hi = *(const v8bf*)(p + 8);
  return cat8(lo, hi);
}

__device__ inline float gelu_f(float v) {
  float u = 0.7978845608028654f * (v + 0.044715f * v * v * v);
  return 0.5f * v * (1.0f + tanhf(u));
}

// ---------------------------------------------------------------- embed
__global__ void k_embed(const int* __restrict__ tok, const float* __restrict__ wte,
                        const float* __restrict__ wpe, float* __restrict__ x) {
  int row = blockIdx.x;
  int t = tok[row];
  for (int c = threadIdx.x; c < D; c += blockDim.x)
    x[(size_t)row * D + c] = wte[(size_t)t * D + c] + wpe[(size_t)row * D + c];
}

// ---------------------------------------------------------------- layernorm -> bf16
__global__ __launch_bounds__(128)
void k_ln(const float* __restrict__ x, const float* __restrict__ w,
          const float* __restrict__ b, __bf16* __restrict__ out) {
  int lane = threadIdx.x & 31;
  int wave = threadIdx.x >> 5;
  int row  = blockIdx.x * 4 + wave;
  const float* xr = x + (size_t)row * D;
  float v[24];
  float s = 0.0f, sq = 0.0f;
#pragma unroll
  for (int i = 0; i < 24; ++i) {
    v[i] = xr[lane + i * 32];
    s += v[i]; sq += v[i] * v[i];
  }
#pragma unroll
  for (int off = 16; off; off >>= 1) {
    s  += __shfl_xor(s,  off, 32);
    sq += __shfl_xor(sq, off, 32);
  }
  float mean = s * (1.0f / 768.0f);
  float var  = sq * (1.0f / 768.0f) - mean * mean;
  float inv  = rsqrtf(var + LN_EPS);
#pragma unroll
  for (int i = 0; i < 24; ++i) {
    int c = lane + i * 32;
    out[(size_t)row * D + c] = (__bf16)((v[i] - mean) * inv * w[c] + b[c]);
  }
}

// ---------------------------------------------------------------- f32 -> bf16 weight convert (8 elems/thread)
__global__ void k_cvt(const float* __restrict__ src, __bf16* __restrict__ dst, long n8) {
  long i = (long)blockIdx.x * blockDim.x + threadIdx.x;
  if (i < n8) {
    f32x8 v = *(const f32x8*)(src + i * 8);
    *(v8bf*)(dst + i * 8) = __builtin_convertvector(v, v8bf);
  }
}

__global__ void k_cvt_pad(const float* __restrict__ src, __bf16* __restrict__ dst,
                          long nvalid8, long ntotal8) {
  long i = (long)blockIdx.x * blockDim.x + threadIdx.x;
  if (i < ntotal8) {
    v8bf o;
    if (i < nvalid8) {
      f32x8 v = *(const f32x8*)(src + i * 8);
      o = __builtin_convertvector(v, v8bf);
    } else {
#pragma unroll
      for (int e = 0; e < 8; ++e) o[e] = (__bf16)0.0f;
    }
    *(v8bf*)(dst + i * 8) = o;
  }
}

// ---------------------------------------------------------------- GEMM: C[M,N] = A[M,K](bf16) @ W[N,K]^T(bf16)
// 32x64 tile per wave, 128x64 per workgroup.
// FLAGS: 1=bias, 2=residual add (R, f32), 4=gelu, 8=bf16 output, 16=col guard (n < Nvalid)
template <int FLAGS>
__global__ __launch_bounds__(128)
void k_gemm(const __bf16* __restrict__ A, const __bf16* __restrict__ W,
            const float* __restrict__ bias, const float* __restrict__ R,
            float* __restrict__ Cf, __bf16* __restrict__ Cb,
            int M, int N, int K, int Nvalid, int ldc) {
  int lane = threadIdx.x & 31;
  int wave = threadIdx.x >> 5;
  int mt = blockIdx.y * 128 + wave * 32;  // 32-row tile per wave
  int nb = blockIdx.x * 64;               // 64-col block per workgroup
  int l15 = lane & 15;
  int hi  = lane >> 4;

  const __bf16* arow0 = A + (size_t)(mt + l15) * K + hi * 16;  // rows mt..mt+15
  const __bf16* arow1 = arow0 + (size_t)16 * K;                // rows mt+16..mt+31
  const __bf16* w0    = W + (size_t)(nb + l15) * K + hi * 16;  // B-frag base

  v8f acc[8];
#pragma unroll
  for (int j = 0; j < 8; ++j) acc[j] = vzero8();

  for (int k0 = 0; k0 < K; k0 += 32) {
    __builtin_prefetch(w0 + k0 + 64, 0, 1);
    v16bf a0 = load_a_bf16(arow0 + k0);
    v16bf a1 = load_a_bf16(arow1 + k0);
    v16bf b0 = *(const v16bf*)(w0 + (size_t)0 * 16 * K + k0);
    v16bf b1 = *(const v16bf*)(w0 + (size_t)1 * 16 * K + k0);
    v16bf b2 = *(const v16bf*)(w0 + (size_t)2 * 16 * K + k0);
    v16bf b3 = *(const v16bf*)(w0 + (size_t)3 * 16 * K + k0);
    acc[0] = __builtin_amdgcn_wmma_f32_16x16x32_bf16(false, a0, false, b0, (short)0, acc[0], false, false);
    acc[1] = __builtin_amdgcn_wmma_f32_16x16x32_bf16(false, a0, false, b1, (short)0, acc[1], false, false);
    acc[2] = __builtin_amdgcn_wmma_f32_16x16x32_bf16(false, a0, false, b2, (short)0, acc[2], false, false);
    acc[3] = __builtin_amdgcn_wmma_f32_16x16x32_bf16(false, a0, false, b3, (short)0, acc[3], false, false);
    acc[4] = __builtin_amdgcn_wmma_f32_16x16x32_bf16(false, a1, false, b0, (short)0, acc[4], false, false);
    acc[5] = __builtin_amdgcn_wmma_f32_16x16x32_bf16(false, a1, false, b1, (short)0, acc[5], false, false);
    acc[6] = __builtin_amdgcn_wmma_f32_16x16x32_bf16(false, a1, false, b2, (short)0, acc[6], false, false);
    acc[7] = __builtin_amdgcn_wmma_f32_16x16x32_bf16(false, a1, false, b3, (short)0, acc[7], false, false);
  }

#pragma unroll
  for (int t = 0; t < 2; ++t) {
    int mrow = mt + t * 16 + hi * 8;
#pragma unroll
    for (int r = 0; r < 8; ++r) {
      int m = mrow + r;
#pragma unroll
      for (int j = 0; j < 4; ++j) {
        float v = acc[t * 4 + j][r];
        int n = nb + j * 16 + l15;
        if (FLAGS & 1) v += bias[n];
        if (FLAGS & 4) v = gelu_f(v);
        if (FLAGS & 2) v += R[(size_t)m * ldc + n];
        if (FLAGS & 16) {
          if (n < Nvalid) Cf[(size_t)m * ldc + n] = v;
        } else if (FLAGS & 8) {
          Cb[(size_t)m * ldc + n] = (__bf16)v;
        } else {
          Cf[(size_t)m * ldc + n] = v;
        }
      }
    }
  }
}

// ---------------------------------------------------------------- flash attention, one wave per (head, 16-query tile)
// qkv is bf16 [L, 3D]; all WMMA operands load directly with no conversion.
__global__ __launch_bounds__(32)
void k_attn(const __bf16* __restrict__ qkv, __bf16* __restrict__ attn) {
  __shared__ alignas(64) __bf16 Pb[16 * 32];   // P tile 16 rows x 32 keys
  __shared__ alignas(64) __bf16 Vt[64 * 32];   // V^T: 64 feats x 32 keys

  int h  = blockIdx.y;
  int m0 = blockIdx.x * 16;
  int lane = threadIdx.x;
  int l15 = lane & 15;
  int hi  = lane >> 4;
  int koff = hi * 8;
  int kb16 = hi * 16;

  // Q fragments (two K=32 chunks of d=64)
  v16bf qa[2];
  {
    const __bf16* qp = qkv + (size_t)(m0 + l15) * 2304 + h * 64 + kb16;
    qa[0] = load_a_bf16(qp);
    qa[1] = load_a_bf16(qp + 32);
  }

  v8f o[4];
#pragma unroll
  for (int j = 0; j < 4; ++j) o[j] = vzero8();
  float rm[8], rl[8];
#pragma unroll
  for (int r = 0; r < 8; ++r) { rm[r] = -3.0e38f; rl[r] = 0.0f; }

  int jend = m0 + 15;
  for (int j0 = 0; j0 <= jend; j0 += 32) {
    // --- scores: two 16x16 tiles covering 32 keys, each K=64 => 2 WMMAs
    v8f s[2];
    s[0] = vzero8(); s[1] = vzero8();
#pragma unroll
    for (int t = 0; t < 2; ++t) {
      int key = j0 + t * 16 + l15;
      const __bf16* kp = qkv + (size_t)key * 2304 + 768 + h * 64 + kb16;
#pragma unroll
      for (int c = 0; c < 2; ++c) {
        v16bf bk = *(const v16bf*)(kp + c * 32);
        s[t] = __builtin_amdgcn_wmma_f32_16x16x32_bf16(false, qa[c], false, bk, (short)0, s[t], false, false);
      }
      s[t] = s[t] * 0.125f;   // 1/sqrt(d), applied in f32 after the dot product
    }
    // --- causal mask
#pragma unroll
    for (int t = 0; t < 2; ++t) {
      int key = j0 + t * 16 + l15;
#pragma unroll
      for (int r = 0; r < 8; ++r) {
        int row = m0 + hi * 8 + r;
        if (key > row) s[t][r] = -3.0e38f;
      }
    }
    // --- stage V^T tile (32 keys x 64 feats) into LDS, pure bf16 copy
    {
      const __bf16* vp = qkv + (size_t)(j0 + lane) * 2304 + 1536 + h * 64;
#pragma unroll
      for (int f8 = 0; f8 < 8; ++f8) {
        v8bf vb = *(const v8bf*)(vp + f8 * 8);
#pragma unroll
        for (int e = 0; e < 8; ++e) Vt[(f8 * 8 + e) * 32 + lane] = vb[e];
      }
    }
    // --- online softmax update, write P to LDS
#pragma unroll
    for (int r = 0; r < 8; ++r) {
      float mx = fmaxf(s[0][r], s[1][r]);
#pragma unroll
      for (int off = 8; off; off >>= 1) mx = fmaxf(mx, __shfl_xor(mx, off, 16));
      float mnew = fmaxf(rm[r], mx);
      float sc = __expf(rm[r] - mnew);
      float p0 = __expf(s[0][r] - mnew);
      float p1 = __expf(s[1][r] - mnew);
      float ps = p0 + p1;
#pragma unroll
      for (int off = 8; off; off >>= 1) ps += __shfl_xor(ps, off, 16);
      rl[r] = rl[r] * sc + ps;
      rm[r] = mnew;
      o[0][r] *= sc; o[1][r] *= sc; o[2][r] *= sc; o[3][r] *= sc;
      int prow = hi * 8 + r;
      Pb[prow * 32 + l15]      = (__bf16)p0;
      Pb[prow * 32 + 16 + l15] = (__bf16)p1;
    }
    __syncthreads();
    // --- O += P(16x32) @ V(32x64): A from Pb, B from Vt, 4 feature tiles
    v16bf pa;
    {
      v8bf plo = *(const v8bf*)(&Pb[l15 * 32 + kb16]);
      v8bf phi = *(const v8bf*)(&Pb[l15 * 32 + kb16 + 8]);
      pa = cat8(plo, phi);
    }
#pragma unroll
    for (int j = 0; j < 4; ++j) {
      v16bf bv = *(const v16bf*)(&Vt[(j * 16 + l15) * 32 + kb16]);
      o[j] = __builtin_amdgcn_wmma_f32_16x16x32_bf16(false, pa, false, bv, (short)0, o[j], false, false);
    }
    __syncthreads();
  }

  // --- normalize and write bf16 output (consumed by proj GEMM)
#pragma unroll
  for (int r = 0; r < 8; ++r) {
    float inv = 1.0f / rl[r];
    int row = m0 + hi * 8 + r;
#pragma unroll
    for (int j = 0; j < 4; ++j)
      attn[(size_t)row * D + h * 64 + j * 16 + l15] = (__bf16)(o[j][r] * inv);
  }
}

// ---------------------------------------------------------------- launch
extern "C" void kernel_launch(void* const* d_in, const int* in_sizes, int n_in,
                              void* d_out, int out_size, void* d_ws, size_t ws_size,
                              hipStream_t stream) {
  const int*   tokens = (const int*)  d_in[0];
  const float* wte    = (const float*)d_in[1];
  const float* wpe    = (const float*)d_in[2];
  const float* ln1_w  = (const float*)d_in[3];
  const float* ln1_b  = (const float*)d_in[4];
  const float* attn_w = (const float*)d_in[5];
  const float* attn_b = (const float*)d_in[6];
  const float* proj_w = (const float*)d_in[7];
  const float* proj_b = (const float*)d_in[8];
  const float* ln2_w  = (const float*)d_in[9];
  const float* ln2_b  = (const float*)d_in[10];
  const float* fc_w   = (const float*)d_in[11];
  const float* fc_b   = (const float*)d_in[12];
  const float* fc2_w  = (const float*)d_in[13];
  const float* fc2_b  = (const float*)d_in[14];
  const float* lnf_w  = (const float*)d_in[15];
  const float* lnf_b  = (const float*)d_in[16];
  const float* head_w = (const float*)d_in[17];
  float* out = (float*)d_out;

  char* ws = (char*)d_ws;
  float*  x    = (float*) (ws + 0);          // [L,D]  f32   3.00 MB
  __bf16* hbuf = (__bf16*)(ws + 3145728);    // [L,D]  bf16  1.50 MB
  __bf16* qkvb = (__bf16*)(ws + 4718592);    // [L,3D] bf16  4.50 MB
  __bf16* attn = (__bf16*)(ws + 9437184);    // [L,D]  bf16  1.50 MB
  __bf16* ffn  = (__bf16*)(ws + 11010048);   // [L,4D] bf16  6.00 MB
  __bf16* wb   = (__bf16*)(ws + 17301504);   // padded head weight, 77.3 MB max

  k_embed<<<LSEQ, 256, 0, stream>>>(tokens, wte, wpe, x);

  for (int b = 0; b < NBLK; ++b) {
    k_ln<<<LSEQ / 4, 128, 0, stream>>>(x, ln1_w + b * D, ln1_b + b * D, hbuf);

    { long n8 = (long)3 * D * D / 8;
      k_cvt<<<(int)((n8 + 255) / 256), 256, 0, stream>>>(attn_w + (size_t)b * 3 * D * D, wb, n8); }
    k_gemm<1 | 8><<<dim3((3 * D) / 64, LSEQ / 128), 128, 0, stream>>>(
        hbuf, wb, attn_b + b * 3 * D, nullptr, nullptr, qkvb, LSEQ, 3 * D, D, 3 * D, 3 * D);

    k_attn<<<dim3(LSEQ / 16, HNUM), 32, 0, stream>>>(qkvb, attn);

    { long n8 = (long)D * D / 8;
      k_cvt<<<(int)((n8 + 255) / 256), 256, 0, stream>>>(proj_w + (size_t)b * D * D, wb, n8); }
    k_gemm<1 | 2><<<dim3(D / 64, LSEQ / 128), 128, 0, stream>>>(
        attn, wb, proj_b + b * D, x, x, nullptr, LSEQ, D, D, D, D);

    k_ln<<<LSEQ / 4, 128, 0, stream>>>(x, ln2_w + b * D, ln2_b + b * D, hbuf);

    { long n8 = (long)4 * D * D / 8;
      k_cvt<<<(int)((n8 + 255) / 256), 256, 0, stream>>>(fc_w + (size_t)b * 4 * D * D, wb, n8); }
    k_gemm<1 | 4 | 8><<<dim3((4 * D) / 64, LSEQ / 128), 128, 0, stream>>>(
        hbuf, wb, fc_b + b * 4 * D, nullptr, nullptr, ffn, LSEQ, 4 * D, D, 4 * D, 4 * D);

    { long n8 = (long)D * 4 * D / 8;
      k_cvt<<<(int)((n8 + 255) / 256), 256, 0, stream>>>(fc2_w + (size_t)b * 4 * D * D, wb, n8); }
    k_gemm<1 | 2><<<dim3(D / 64, LSEQ / 128), 128, 0, stream>>>(
        ffn, wb, fc2_b + b * D, x, x, nullptr, LSEQ, D, 4 * D, D, D);
  }

  k_ln<<<LSEQ / 4, 128, 0, stream>>>(x, lnf_w, lnf_b, hbuf);

  { long nt8 = (long)VPAD * D / 8, nv8 = (long)VOCAB * D / 8;
    k_cvt_pad<<<(int)((nt8 + 255) / 256), 256, 0, stream>>>(head_w, wb, nv8, nt8); }
  k_gemm<16><<<dim3(VPAD / 64, LSEQ / 128), 128, 0, stream>>>(
      hbuf, wb, nullptr, nullptr, out, nullptr, LSEQ, VPAD, D, VOCAB, VOCAB);
}
